// LSTM_LM_18442589569428
// MI455X (gfx1250) — matrix operand order repaired and checked by reading, not compiled, not run on hardware
//
#include <hip/hip_runtime.h>

// ---------------------------------------------------------------------------
// Types for CDNA5 WMMA (wave32): v_wmma_f32_16x16x32_bf16
// ---------------------------------------------------------------------------
typedef __bf16 bf16;
typedef __attribute__((ext_vector_type(16))) __bf16 v16bf;
typedef __attribute__((ext_vector_type(8)))  __bf16 v8bf;
typedef __attribute__((ext_vector_type(4)))  __bf16 v4bf;
typedef __attribute__((ext_vector_type(8)))  float  v8f;
typedef __attribute__((ext_vector_type(4)))  float  v4f;

union FragBF {
    v16bf v;
    v8bf  h[2];
};

// Model constants
#define LV 32000
#define LE 128
#define LH 256
#define LB 16
#define LT 256
#define LM (LB * LT)       // 4096 rows
#define LG (4 * LH)        // 1024 gate columns

// ---------------------------------------------------------------------------
// Fast activations: avoid the IEEE v_div_scale/div_fmas/div_fixup slow path
// (the recurrence is latency-bound: 512 sequential steps). v_rcp_f32 + v_exp.
// tanh(x) = 1 - 2/(1 + e^{2x}) saturates correctly via exp->inf / exp->0.
// ---------------------------------------------------------------------------
__device__ __forceinline__ float fast_sigmoid(float x) {
    return __builtin_amdgcn_rcpf(1.f + __expf(-x));
}
__device__ __forceinline__ float fast_tanh(float x) {
    return 1.f - 2.f * __builtin_amdgcn_rcpf(1.f + __expf(2.f * x));
}

// ---------------------------------------------------------------------------
// fp32 -> bf16 conversion, 4 elements/thread. Source is read exactly once ->
// non-temporal load; destination is re-read by GEMMs -> regular (RT) store.
// ---------------------------------------------------------------------------
__global__ void k_cvt_bf16_x4(const float* __restrict__ in, bf16* __restrict__ out, int n4) {
    int i = blockIdx.x * blockDim.x + threadIdx.x;
    if (i < n4) {
        v4f a = __builtin_nontemporal_load(reinterpret_cast<const v4f*>(in) + i);
        v4bf b;
        b.x = (bf16)a.x; b.y = (bf16)a.y; b.z = (bf16)a.z; b.w = (bf16)a.w;
        reinterpret_cast<v4bf*>(out)[i] = b;
    }
}

// bias = b_ih + b_hh
__global__ void k_bias_add(const float* __restrict__ a, const float* __restrict__ b,
                           float* __restrict__ out, int n) {
    int i = blockIdx.x * blockDim.x + threadIdx.x;
    if (i < n) out[i] = a[i] + b[i];
}

// Embedding gather -> bf16, rows m = b*T + t (matches [B,T,E] layout).
// 4 elements/thread: 32 threads per embedding row (LE = 128).
__global__ void k_embed(const int* __restrict__ x, const float* __restrict__ ew,
                        bf16* __restrict__ out) {
    int i = blockIdx.x * blockDim.x + threadIdx.x;   // over LM*LE/4
    int m  = i >> 5;                                  // token row
    int e4 = (i & 31);                                // 4-elem chunk in row
    v4f a = *(reinterpret_cast<const v4f*>(ew + (size_t)x[m] * LE) + e4);
    v4bf b;
    b.x = (bf16)a.x; b.y = (bf16)a.y; b.z = (bf16)a.z; b.w = (bf16)a.w;
    reinterpret_cast<v4bf*>(out)[i] = b;
}

// ---------------------------------------------------------------------------
// GEMM: C[M,N] (f32) = A[M,K] (bf16, row-major) x W[N,K]^T (bf16, row-major)
//                      + bias[N]
// Block: 256 threads (8 waves). Tile: 64(M) x 128(N) x 32(K).
// NT_STORE=true for the logits GEMM: the 524 MB output stream is never
// re-read, so bypass-hint it to keep fc_w/h2 resident in the 192 MB L2.
// ---------------------------------------------------------------------------
#define GBM 64
#define GBN 128
#define GBK 32

template <bool NT_STORE>
__global__ __launch_bounds__(256) void k_gemm_bf16(
    const bf16* __restrict__ A, const bf16* __restrict__ W,
    const float* __restrict__ bias, float* __restrict__ C,
    int M, int N, int K) {

    __shared__ __align__(64) bf16 As[GBM][GBK];
    __shared__ __align__(64) bf16 Ws[GBN][GBK];

    const int tid  = threadIdx.x;
    const int lane = tid & 31;
    const int wave = tid >> 5;
    const int bm   = blockIdx.y * GBM;
    const int bn   = blockIdx.x * GBN;
    const int wm   = wave & 3;        // M subtile 0..3
    const int wn   = wave >> 2;       // N group 0..1 (64 cols each)
    const int r    = lane & 15;
    const int half = lane >> 4;
    const int kb   = half * 8;        // unused: ISA-table K base; frags here load K at 16*half

    v8f acc[4] = {};

    for (int kk = 0; kk < K; kk += GBK) {
        // Stage A tile: 64x32 bf16 = 256 chunks of 16B, one per thread
        {
            int row = tid >> 2;
            int c8  = tid & 3;
            *reinterpret_cast<v8bf*>(&As[row][c8 * 8]) =
                *reinterpret_cast<const v8bf*>(A + (size_t)(bm + row) * K + kk + c8 * 8);
        }
        // Stage W tile: 128x32 bf16 = 512 chunks, two per thread
        #pragma unroll
        for (int it = 0; it < 2; ++it) {
            int chunk = tid + it * 256;
            int row   = chunk >> 2;
            int c8    = chunk & 3;
            *reinterpret_cast<v8bf*>(&Ws[row][c8 * 8]) =
                *reinterpret_cast<const v8bf*>(W + (size_t)(bn + row) * K + kk + c8 * 8);
        }
        __syncthreads();

        // A fragment: lane holds row (wm*16 + r), 16 contiguous K at 16*half
        FragBF a;
        a.v = *reinterpret_cast<const v16bf*>(&As[wm * 16 + r][half * 16]);

        #pragma unroll
        for (int nt = 0; nt < 4; ++nt) {
            // B fragment: lane -> column (W row), 16 contiguous K at 16*half
            FragBF b;
            const int wrow = wn * 64 + nt * 16 + r;
            b.v = *reinterpret_cast<const v16bf*>(&Ws[wrow][half * 16]);
            acc[nt] = __builtin_amdgcn_wmma_f32_16x16x32_bf16(
                false, a.v, false, b.v, (short)0, acc[nt], false, false);
        }
        __syncthreads();
    }

    // Epilogue: D layout f32 16x16 -> (M = j + 8*half, N = r)
    #pragma unroll
    for (int nt = 0; nt < 4; ++nt) {
        const int col = bn + wn * 64 + nt * 16 + r;
        const float bv = bias ? bias[col] : 0.f;
        #pragma unroll
        for (int j = 0; j < 8; ++j) {
            const int row = bm + wm * 16 + j + half * 8;
            float* p = &C[(size_t)row * N + col];
            float  v = acc[nt][j] + bv;
            if (NT_STORE) __builtin_nontemporal_store(v, p);
            else          *p = v;
        }
    }
}

// ---------------------------------------------------------------------------
// LSTM recurrence: persistent single workgroup, 1024 threads = 32 waves.
// State h (bf16) and c (f32) LDS-resident. Per step:
//   gates[16,1024] = h @ w_hh^T + x_proj[t]  (WMMA, 2 gate tiles per wave)
//   then i/f/g/o nonlinearity + c/h update across all threads.
// The compiler keeps most Whh fragments register-resident across t (spilling
// the overflow to L2-cached scratch), which beats re-streaming all 512 KB.
// ---------------------------------------------------------------------------
__global__ __launch_bounds__(1024) void k_lstm_seq(
    const float* __restrict__ xp,    // [LM, LG], row m = b*T + t (bias pre-added)
    const bf16*  __restrict__ Whh,   // [LG, LH] bf16 row-major
    bf16* __restrict__ h_out) {      // [LM, LH] bf16, row m = b*T + t

    __shared__ __align__(64) bf16 h_s[LB][LH];     // 8 KB
    __shared__ float c_s[LB][LH];                  // 16 KB
    __shared__ float g_s[LB][LG];                  // 64 KB

    const int tid  = threadIdx.x;
    const int lane = tid & 31;
    const int wave = tid >> 5;       // 0..31
    const int r    = lane & 15;
    const int half = lane >> 4;
    const int kb   = half * 8;

    for (int e = tid; e < LB * LH; e += 1024) {
        h_s[e >> 8][e & 255] = (bf16)0.f;
        c_s[e >> 8][e & 255] = 0.f;
    }
    __syncthreads();

    for (int t = 0; t < LT; ++t) {
        // --- gate GEMM: each wave owns gate tiles (wave*2) and (wave*2+1) ---
        #pragma unroll
        for (int i = 0; i < 2; ++i) {
            const int n0 = (wave * 2 + i) * 16;
            v8f acc = {};
            #pragma unroll
            for (int k8 = 0; k8 < 8; ++k8) {
                const int kk = k8 * 32;
                FragBF a, b;
                a.v = *reinterpret_cast<const v16bf*>(&h_s[r][kk + half * 16]);
                const bf16* wrow = Whh + (size_t)(n0 + r) * LH + kk;
                b.v = *reinterpret_cast<const v16bf*>(wrow + half * 16);
                acc = __builtin_amdgcn_wmma_f32_16x16x32_bf16(
                    false, a.v, false, b.v, (short)0, acc, false, false);
            }
            #pragma unroll
            for (int j = 0; j < 8; ++j) {
                const int bb = j + half * 8;                      // batch row
                g_s[bb][n0 + r] = acc[j] +
                    xp[((size_t)bb * LT + t) * LG + n0 + r];
            }
        }
        __syncthreads();

        // --- elementwise i/f/g/o -> c,h update: 4096 elems / 1024 threads ---
        #pragma unroll
        for (int p = 0; p < 4; ++p) {
            const int e  = tid + p * 1024;
            const int bb = e >> 8;
            const int hc = e & 255;
            float gi = fast_sigmoid(g_s[bb][hc]);
            float gf = fast_sigmoid(g_s[bb][LH + hc]);
            float gg = fast_tanh   (g_s[bb][2 * LH + hc]);
            float go = fast_sigmoid(g_s[bb][3 * LH + hc]);
            float c = gf * c_s[bb][hc] + gi * gg;
            c_s[bb][hc] = c;
            float h = go * fast_tanh(c);
            h_s[bb][hc] = (bf16)h;
            h_out[((size_t)bb * LT + t) * LH + hc] = (bf16)h;
        }
        __syncthreads();
    }
}

// ---------------------------------------------------------------------------
// Host-side orchestration
// ---------------------------------------------------------------------------
extern "C" void kernel_launch(void* const* d_in, const int* in_sizes, int n_in,
                              void* d_out, int out_size, void* d_ws, size_t ws_size,
                              hipStream_t stream) {
    (void)in_sizes; (void)n_in; (void)out_size; (void)ws_size;

    const int*   x       = (const int*)  d_in[0];
    const float* embed_w = (const float*)d_in[1];
    const float* w_ih0   = (const float*)d_in[2];
    const float* w_hh0   = (const float*)d_in[3];
    const float* b_ih0   = (const float*)d_in[4];
    const float* b_hh0   = (const float*)d_in[5];
    const float* w_ih1   = (const float*)d_in[6];
    const float* w_hh1   = (const float*)d_in[7];
    const float* b_ih1   = (const float*)d_in[8];
    const float* b_hh1   = (const float*)d_in[9];
    const float* fc_w    = (const float*)d_in[10];
    const float* fc_b    = (const float*)d_in[11];
    float* logits = (float*)d_out;

    // Workspace carve-out (256B aligned)
    char* ws = (char*)d_ws;
    size_t off = 0;
    auto alloc = [&](size_t bytes) -> void* {
        void* p = ws + off;
        off = (off + bytes + 255) & ~(size_t)255;
        return p;
    };
    bf16*  emb_b   = (bf16*) alloc((size_t)LM * LE * 2);      // 1 MB
    bf16*  wih0_b  = (bf16*) alloc((size_t)LG * LE * 2);      // 256 KB
    bf16*  whh0_b  = (bf16*) alloc((size_t)LG * LH * 2);      // 512 KB
    bf16*  wih1_b  = (bf16*) alloc((size_t)LG * LH * 2);      // 512 KB
    bf16*  whh1_b  = (bf16*) alloc((size_t)LG * LH * 2);      // 512 KB
    bf16*  fcw_b   = (bf16*) alloc((size_t)LV * LH * 2);      // 16 MB
    float* bias0   = (float*)alloc((size_t)LG * 4);
    float* bias1   = (float*)alloc((size_t)LG * 4);
    float* xp0     = (float*)alloc((size_t)LM * LG * 4);      // 16 MB
    float* xp1     = (float*)alloc((size_t)LM * LG * 4);      // 16 MB
    bf16*  h1_b    = (bf16*) alloc((size_t)LM * LH * 2);      // 2 MB
    bf16*  h2_b    = (bf16*) alloc((size_t)LM * LH * 2);      // 2 MB

    const int TB = 256;
    auto blocks = [](int n, int tb) { return (n + tb - 1) / tb; };

    // Weight / bias preparation (x4 vectorized; all counts divisible by 4)
    k_cvt_bf16_x4<<<blocks(LG * LE / 4, TB), TB, 0, stream>>>(w_ih0, wih0_b, LG * LE / 4);
    k_cvt_bf16_x4<<<blocks(LG * LH / 4, TB), TB, 0, stream>>>(w_hh0, whh0_b, LG * LH / 4);
    k_cvt_bf16_x4<<<blocks(LG * LH / 4, TB), TB, 0, stream>>>(w_ih1, wih1_b, LG * LH / 4);
    k_cvt_bf16_x4<<<blocks(LG * LH / 4, TB), TB, 0, stream>>>(w_hh1, whh1_b, LG * LH / 4);
    k_cvt_bf16_x4<<<blocks(LV * LH / 4, TB), TB, 0, stream>>>(fc_w,  fcw_b,  LV * LH / 4);
    k_bias_add<<<blocks(LG, TB), TB, 0, stream>>>(b_ih0, b_hh0, bias0, LG);
    k_bias_add<<<blocks(LG, TB), TB, 0, stream>>>(b_ih1, b_hh1, bias1, LG);

    // Embedding gather (bf16)
    k_embed<<<blocks(LM * LE / 4, TB), TB, 0, stream>>>(x, embed_w, emb_b);

    // Layer 0: input projection + recurrence
    k_gemm_bf16<false><<<dim3(LG / GBN, LM / GBM), TB, 0, stream>>>(
        emb_b, wih0_b, bias0, xp0, LM, LG, LE);
    k_lstm_seq<<<1, 1024, 0, stream>>>(xp0, whh0_b, h1_b);

    // Layer 1: input projection + recurrence
    k_gemm_bf16<false><<<dim3(LG / GBN, LM / GBM), TB, 0, stream>>>(
        h1_b, wih1_b, bias1, xp1, LM, LG, LH);
    k_lstm_seq<<<1, 1024, 0, stream>>>(xp1, whh1_b, h2_b);

    // FC logits: HBM-store-bound (524 MB out) -> non-temporal stores
    k_gemm_bf16<true><<<dim3(LV / GBN, LM / GBM), TB, 0, stream>>>(
        h2_b, fcw_b, fc_b, logits, LM, LV, LH);
}
